// AnyPrecisionLinear_2757369004801
// MI455X (gfx1250) — matrix operand order repaired and checked by reading, not compiled, not run on hardware
//
#include <hip/hip_runtime.h>

typedef __attribute__((ext_vector_type(16))) _Float16 v16h;
typedef __attribute__((ext_vector_type(8)))  _Float16 v8h;
typedef __attribute__((ext_vector_type(2)))  _Float16 v2h;
typedef __attribute__((ext_vector_type(8)))  float    v8f;

#define KDIM  4096
#define NDIM  4096
#define WORDS 128          // KDIM / 32
#define QBITS 8

// ---------------------------------------------------------------------------
// Kernel 1: bit-plane LUT dequantization.  qweight is (8, 4096, 128) int32,
// bit-plane packed with plane 0 as the MSB of the 8-bit index.  One block per
// output row n; the 256-entry fp16 LUT row is staged in LDS; each of the 128
// threads decodes one 32-bit word -> 32 fp16 weights -> 64B vector stores.
// ---------------------------------------------------------------------------
__global__ __launch_bounds__(128)
void anyprec_dequant_kernel(const int* __restrict__ qw,
                            const _Float16* __restrict__ lut,
                            _Float16* __restrict__ W)
{
    __shared__ _Float16 slut[256];
    const int n = blockIdx.x;
    const int t = threadIdx.x;

    ((v2h*)slut)[t] = ((const v2h*)(lut + (size_t)n * 256))[t];
    __syncthreads();

    unsigned pl[QBITS];
#pragma unroll
    for (int p = 0; p < QBITS; ++p)
        pl[p] = (unsigned)qw[((size_t)p * NDIM + n) * WORDS + t];

    _Float16 w[32];
#pragma unroll
    for (int b = 0; b < 32; ++b) {
        int idx = 0;
#pragma unroll
        for (int p = 0; p < QBITS; ++p)
            idx = (idx << 1) | (int)((pl[p] >> b) & 1u);   // plane 0 = MSB
        w[b] = slut[idx];
    }

    v8h* dst = (v8h*)(W + (size_t)n * KDIM + t * 32);
#pragma unroll
    for (int c = 0; c < 4; ++c) {
        v8h v;
#pragma unroll
        for (int e = 0; e < 8; ++e) v[e] = w[c * 8 + e];
        dst[c] = v;
    }
}

// ---------------------------------------------------------------------------
// Kernel 2: WMMA fp16 GEMM  out[m][n] = sum_k X[m][k] * W[n][k] + bias[n]
// Block = 256 threads (8 wave32), C tile 128(M) x 128(N).
// Wave grid 2(M) x 4(N); each wave computes 4x2 = 8 tiles of 16x16
// via v_wmma_f32_16x16x32_f16 with f32 accumulation.
//
// Software pipeline: fragments for K-step t+1 are loaded into a second
// register buffer BEFORE the WMMAs of K-step t, so the wave only waits for
// the older half of its outstanding loads (s_wait_loadcnt != 0) while the
// matrix pipe runs.
//
// Fragment layouts used here, contiguous k = 16*half + s (lane = (idx16, half)):
//   A (16x32 f16): halves [half*16 .. +16) of row m
//   B (32x16 f16): halves [half*16 .. +16) of column n  (== row n of W)
//   C/D (16x16 f32): vgpr v holds row v + 8*half, column = lane & 15
// ---------------------------------------------------------------------------
#define BM 128
#define BN 128
#define MW 4   // 16-wide M tiles per wave
#define NW 2   // 16-wide N tiles per wave
#define KT 32  // K per WMMA

struct Frags {
    v16h a[MW];
    v16h b[NW];
};

__device__ __forceinline__ void load_frags(Frags& f,
                                           const _Float16* const* rowA,
                                           const _Float16* const* rowB,
                                           int k, int aoff0, int aoff1, int boff)
{
    union { v16h v; v8h h[2]; } t;
#pragma unroll
    for (int j = 0; j < NW; ++j) {
        t.h[0] = *(const v8h*)(rowB[j] + k + boff);
        t.h[1] = *(const v8h*)(rowB[j] + k + boff + 8);
        f.b[j] = t.v;
    }
#pragma unroll
    for (int i = 0; i < MW; ++i) {
        t.h[0] = *(const v8h*)(rowA[i] + k + aoff0);
        t.h[1] = *(const v8h*)(rowA[i] + k + aoff1);
        f.a[i] = t.v;
    }
}

// reuse_a / reuse_b must be literal constants (frontend ICE check), so the
// 4x2 MMA block is unrolled manually.  j-outer / i-inner: consecutive WMMAs
// share the B operand; reuse_b=true on all but the last user of each B.
#define WMMA_ONE(i, j, RB)                                                    \
    acc[i][j] = __builtin_amdgcn_wmma_f32_16x16x32_f16(                       \
        false, f.a[i], false, f.b[j], (short)0, acc[i][j], false, RB)

__device__ __forceinline__ void mma_step(const Frags& f, v8f acc[MW][NW])
{
    WMMA_ONE(0, 0, true);
    WMMA_ONE(1, 0, true);
    WMMA_ONE(2, 0, true);
    WMMA_ONE(3, 0, false);   // next instruction switches B
    WMMA_ONE(0, 1, true);
    WMMA_ONE(1, 1, true);
    WMMA_ONE(2, 1, true);
    WMMA_ONE(3, 1, false);
}

__global__ __launch_bounds__(256)
void anyprec_wmma_gemm_kernel(const _Float16* __restrict__ X,
                              const _Float16* __restrict__ W,
                              const _Float16* __restrict__ bias,
                              float* __restrict__ out)
{
    const int tid  = threadIdx.x;
    const int lane = tid & 31;
    const int wave = tid >> 5;
    const int wm   = wave & 1;    // 0..1  -> 64 rows each
    const int wn   = wave >> 1;   // 0..3  -> 32 cols each
    const int m16  = lane & 15;
    const int half = lane >> 4;

    const int mBase = blockIdx.y * BM + wm * (MW * 16);
    const int nBase = blockIdx.x * BN + wn * (NW * 16);

    const _Float16* rowA[MW];
    const _Float16* rowB[NW];
#pragma unroll
    for (int i = 0; i < MW; ++i)
        rowA[i] = X + (size_t)(mBase + i * 16 + m16) * KDIM;
#pragma unroll
    for (int j = 0; j < NW; ++j)
        rowB[j] = W + (size_t)(nBase + j * 16 + m16) * KDIM;

    const int aoff0 = half * 16;       // in halves
    const int aoff1 = half * 16 + 8;
    const int boff  = half * 16;

    v8f acc[MW][NW] = {};

    Frags f0, f1;
    load_frags(f0, rowA, rowB, 0, aoff0, aoff1, boff);

#pragma unroll 1
    for (int k = 0; k < KDIM; k += 2 * KT) {
        // prefetch far-ahead cachelines for the first A/B rows of this wave
        if (k + 512 < KDIM) {                        // global_prefetch_b8
            __builtin_prefetch(rowA[0] + k + 512, 0, 1);
            __builtin_prefetch(rowB[0] + k + 512, 0, 1);
        }
        load_frags(f1, rowA, rowB, k + KT, aoff0, aoff1, boff);
        mma_step(f0, acc);                           // overlaps f1 loads
        if (k + 2 * KT < KDIM)
            load_frags(f0, rowA, rowB, k + 2 * KT, aoff0, aoff1, boff);
        mma_step(f1, acc);                           // overlaps f0 loads
    }

    // Epilogue: fold fp16 bias, write fp32 output (row-major M x N).
#pragma unroll
    for (int j = 0; j < NW; ++j) {
        const int   nc = nBase + j * 16 + m16;
        const float bv = (float)bias[nc];
#pragma unroll
        for (int i = 0; i < MW; ++i) {
            const int mrow = mBase + i * 16 + half * 8;
#pragma unroll
            for (int v = 0; v < 8; ++v)
                out[(size_t)(mrow + v) * NDIM + nc] = acc[i][j][v] + bv;
        }
    }
}

// ---------------------------------------------------------------------------
// Launch: dequant (4096 blocks) then GEMM (32 x 64 blocks) on `stream`.
// d_ws holds the dequantized fp16 weight matrix: 4096*4096*2 = 32 MB.
// ---------------------------------------------------------------------------
extern "C" void kernel_launch(void* const* d_in, const int* in_sizes, int n_in,
                              void* d_out, int out_size, void* d_ws, size_t ws_size,
                              hipStream_t stream)
{
    const _Float16* x    = (const _Float16*)d_in[0];   // (4,2048,4096) fp16
    const int*      qw   = (const int*)d_in[1];        // (8,4096,128) int32
    const _Float16* lut  = (const _Float16*)d_in[2];   // (4096,256) fp16
    const _Float16* bias = (const _Float16*)d_in[3];   // (4096,) fp16
    float*          out  = (float*)d_out;              // (8192,4096) fp32
    _Float16*       W    = (_Float16*)d_ws;            // 32 MB scratch

    const int Mtot = in_sizes[0] / KDIM;               // 8192

    anyprec_dequant_kernel<<<NDIM, 128, 0, stream>>>(qw, lut, W);

    dim3 grid(NDIM / BN, Mtot / BM);
    anyprec_wmma_gemm_kernel<<<grid, 256, 0, stream>>>(x, W, bias, out);
}
